// Memory_Autograd_49134425866264
// MI455X (gfx1250) — compile-verified, matrix operand order checked by reading, not run on hardware
//
#include <hip/hip_runtime.h>
#include <math.h>

// ---------------------------------------------------------------------------
// Memory-attention forward on MI455X (gfx1250), wave32 + WMMA bf16.
//
// r = a_q@V - tril(a_q a_k^T, -1) @ (a_k@V - v)
// Chunked causal linear attention with [M,DV] running state (~45 GFLOP vs
// 550 GFLOP naive). All GEMMs on v_wmma_f32_16x16x32_bf16; LDS tiles are in
// fragment order (reads = 2x ds_load_b128); tile copies are 16B->16B and use
// GLOBAL_LOAD_ASYNC_TO_LDS_B128 when the toolchain exposes it.
// ---------------------------------------------------------------------------

typedef __attribute__((ext_vector_type(16))) __bf16 v16bf;
typedef __attribute__((ext_vector_type(8)))  float  v8f;
typedef __attribute__((ext_vector_type(4)))  int    v4i;

#define T_TOK  8192
#define M_SLOT 4096
#define DD     64
#define CHUNK  256
#define NCHUNK (T_TOK / CHUNK)   // 32

#define AS1 __attribute__((address_space(1)))
#define AS3 __attribute__((address_space(3)))

#if defined(__has_builtin)
# if __has_builtin(__builtin_amdgcn_global_load_async_to_lds_b128)
#  define ASYNC_OK 1
# endif
#endif
#ifndef ASYNC_OK
# define ASYNC_OK 0
#endif

// ------------- fragment-order LDS addressing (wave32 16x16x32 bf16) --------
__device__ __forceinline__ int fa_idx(int r, int k, int KT) {
  int rt = r >> 4, kt = k >> 5, kl = k & 31;
  int hi = (kl >> 3) & 1;
  int lane = hi * 16 + (r & 15);
  int e = (kl & 7) + ((kl >> 4) << 3);
  return ((rt * KT + kt) * 32 + lane) * 16 + e;
}
__device__ __forceinline__ int fb_idx(int k, int n, int NT) {
  int kt = k >> 5, nt = n >> 4, kl = k & 31;
  int hi = (kl >> 3) & 1;
  int lane = hi * 16 + (n & 15);
  int e = (kl & 7) + ((kl >> 4) << 3);
  return ((kt * NT + nt) * 32 + lane) * 16 + e;
}
__device__ __forceinline__ v16bf frag_ld(const __bf16* buf, int sub, int lane) {
  return ((const v16bf*)buf)[sub * 32 + lane];
}
__device__ __forceinline__ v8f wmma_bf16(const v16bf& a, const v16bf& b, const v8f& c) {
  return __builtin_amdgcn_wmma_f32_16x16x32_bf16(false, a, false, b, (short)0, c, false, false);
}
#define V8F_ZERO {0.f,0.f,0.f,0.f,0.f,0.f,0.f,0.f}

union U16x8 { uint4 u; __bf16 h[8]; };

// 16B global -> 16B LDS copy; async when available.
__device__ __forceinline__ void cp16(__bf16* lds_dst, const __bf16* gsrc) {
#if ASYNC_OK
  __builtin_amdgcn_global_load_async_to_lds_b128(
      (AS1 v4i*)gsrc, (AS3 v4i*)lds_dst, 0, 0);
#else
  *(uint4*)lds_dst = *(const uint4*)gsrc;
#endif
}
__device__ __forceinline__ void wait_async() {
#if ASYNC_OK
# if defined(__has_builtin) && __has_builtin(__builtin_amdgcn_s_wait_asynccnt)
  __builtin_amdgcn_s_wait_asynccnt(0);
# else
  asm volatile("s_wait_asynccnt 0" ::: "memory");
# endif
#endif
}

// ---- global -> fragment-ordered LDS loaders -------------------------------
// A tile ROWS x (32*KT), src row-major (contiguous k): 16B -> 16B.
template<int ROWS, int KT>
__device__ __forceinline__ void ldA_rm(__bf16* dst, const __bf16* src, int stride, int tid) {
  for (int idx = tid; idx < ROWS * KT * 4; idx += 256) {
    int r  = idx / (KT * 4);
    int k0 = (idx % (KT * 4)) * 8;
    cp16(dst + fa_idx(r, k0, KT), src + (size_t)r * stride + k0);
  }
}
// A tile, src element (r,k) at src[k*stride + r] (contiguous r): scatter.
template<int ROWS, int KT>
__device__ __forceinline__ void ldA_cm(__bf16* dst, const __bf16* src, int stride, int tid) {
  for (int idx = tid; idx < (KT * 32) * (ROWS / 8); idx += 256) {
    int k  = idx / (ROWS / 8);
    int r0 = (idx % (ROWS / 8)) * 8;
    U16x8 u; u.u = *(const uint4*)(src + (size_t)k * stride + r0);
#pragma unroll
    for (int t = 0; t < 8; ++t) dst[fa_idx(r0 + t, k, KT)] = u.h[t];
  }
}
// B tile (32*KT) x (16*NT), src row-major src[k*stride + n] (contiguous n): scatter.
template<int KT, int NT>
__device__ __forceinline__ void ldB_rm(__bf16* dst, const __bf16* src, int stride, int tid) {
  for (int idx = tid; idx < KT * 32 * NT * 2; idx += 256) {
    int k  = idx / (NT * 2);
    int n0 = (idx % (NT * 2)) * 8;
    U16x8 u; u.u = *(const uint4*)(src + (size_t)k * stride + n0);
#pragma unroll
    for (int t = 0; t < 8; ++t) dst[fb_idx(k, n0 + t, NT)] = u.h[t];
  }
}
// B^T tile: element (k,n) at src[n*stride + k] (contiguous k): 16B -> 16B.
template<int KT, int NT>
__device__ __forceinline__ void ldB_tr(__bf16* dst, const __bf16* src, int stride, int tid) {
  for (int idx = tid; idx < NT * 16 * KT * 4; idx += 256) {
    int n  = idx / (KT * 4);
    int k0 = (idx % (KT * 4)) * 8;
    cp16(dst + fb_idx(k0, n, NT), src + (size_t)n * stride + k0);
  }
}

// ---------------------------------------------------------------------------
// 1) elementwise f32 -> bf16
__global__ void k_cvt_bf16(const float* __restrict__ src, __bf16* __restrict__ dst, int n) {
  int i = blockIdx.x * blockDim.x + threadIdx.x;
  int stride = gridDim.x * blockDim.x;
  for (; i < n; i += stride) dst[i] = (__bf16)src[i];
}

// ---------------------------------------------------------------------------
// 2) P[rb:rb+64, cb:cb+64] = bf16( scale * X @ Km^T ).  X:[T,64] Km:[M,64]
__global__ void k_logits(const __bf16* __restrict__ X, const __bf16* __restrict__ Km,
                         __bf16* __restrict__ P, float scale) {
  __shared__ __align__(32) __bf16 sX[4096];
  __shared__ __align__(32) __bf16 sK[4096];
  int tid = threadIdx.x, lane = tid & 31, wv = tid >> 5;
  int rb = blockIdx.x * 64, cb = blockIdx.y * 64;

  ldA_rm<64, 2>(sX, X + (size_t)rb * DD, DD, tid);
  ldB_tr<2, 4>(sK, Km + (size_t)cb * DD, DD, tid);   // B[k][n] = Km[n][k]
  wait_async();
  __syncthreads();

#pragma unroll
  for (int nt = 0; nt < 4; ++nt) {
    v8f acc = V8F_ZERO;
#pragma unroll
    for (int kt = 0; kt < 2; ++kt)
      acc = wmma_bf16(frag_ld(sX, wv * 2 + kt, lane), frag_ld(sK, kt * 4 + nt, lane), acc);
    int col   = cb + nt * 16 + (lane & 15);
    int rbase = rb + wv * 16 + ((lane >> 4) << 3);
#pragma unroll
    for (int j = 0; j < 8; ++j)
      P[(size_t)(rbase + j) * M_SLOT + col] = (__bf16)(acc[j] * scale);
  }
}

// ---------------------------------------------------------------------------
// 3) in-place row softmax over M=4096; one block per row.
__global__ void k_softmax(__bf16* __restrict__ P) {
  __shared__ float red[8];
  int tid = threadIdx.x;
  __bf16* prow = P + (size_t)blockIdx.x * M_SLOT;

  float mx = -3.0e38f;
  for (int c = tid; c < M_SLOT; c += 256) mx = fmaxf(mx, (float)prow[c]);
#pragma unroll
  for (int o = 16; o > 0; o >>= 1) mx = fmaxf(mx, __shfl_xor(mx, o, 32));
  if ((tid & 31) == 0) red[tid >> 5] = mx;
  __syncthreads();
  mx = red[0];
#pragma unroll
  for (int w = 1; w < 8; ++w) mx = fmaxf(mx, red[w]);
  __syncthreads();

  float sum = 0.f;
  for (int c = tid; c < M_SLOT; c += 256) sum += __expf((float)prow[c] - mx);
#pragma unroll
  for (int o = 16; o > 0; o >>= 1) sum += __shfl_xor(sum, o, 32);
  if ((tid & 31) == 0) red[tid >> 5] = sum;
  __syncthreads();
  sum = 0.f;
#pragma unroll
  for (int w = 0; w < 8; ++w) sum += red[w];
  float inv = 1.0f / sum;

  for (int c = tid; c < M_SLOT; c += 256)
    prow[c] = (__bf16)(__expf((float)prow[c] - mx) * inv);
}

// ---------------------------------------------------------------------------
// 4) out = P @ Vb (+ sign*src).  outF: [T,64] f32 (opt). outT: [64,T] bf16
//    TRANSPOSED (opt) -- transposed store is contiguous per column (uint4).
__global__ void k_pv(const __bf16* __restrict__ P, const __bf16* __restrict__ Vb,
                     const float* __restrict__ src, float sign,
                     float* __restrict__ outF, __bf16* __restrict__ outT) {
  __shared__ __align__(32) __bf16 sP[4096];
  __shared__ __align__(32) __bf16 sV[4096];
  int tid = threadIdx.x, lane = tid & 31, wv = tid >> 5;
  int rb = blockIdx.x * 64;

  v8f acc[4];
  const v8f vz = V8F_ZERO;
#pragma unroll
  for (int t = 0; t < 4; ++t) acc[t] = vz;

  for (int m0 = 0; m0 < M_SLOT; m0 += 64) {
    __syncthreads();
    ldA_rm<64, 2>(sP, P + (size_t)rb * M_SLOT + m0, M_SLOT, tid);
    ldB_rm<2, 4>(sV, Vb + (size_t)m0 * DD, DD, tid);
    if (m0 + 64 < M_SLOT)
      __builtin_prefetch(P + (size_t)(rb + (tid & 63)) * M_SLOT + m0 + 64, 0, 3);
    wait_async();
    __syncthreads();
#pragma unroll
    for (int nt = 0; nt < 4; ++nt)
#pragma unroll
      for (int kt = 0; kt < 2; ++kt)
        acc[nt] = wmma_bf16(frag_ld(sP, wv * 2 + kt, lane), frag_ld(sV, kt * 4 + nt, lane), acc[nt]);
  }

#pragma unroll
  for (int nt = 0; nt < 4; ++nt) {
    int col   = nt * 16 + (lane & 15);
    int rbase = rb + wv * 16 + ((lane >> 4) << 3);
    U16x8 u;
#pragma unroll
    for (int j = 0; j < 8; ++j) {
      float val = acc[nt][j];
      if (src) val += sign * src[(size_t)(rbase + j) * DD + col];
      if (outF) outF[(size_t)(rbase + j) * DD + col] = val;
      u.h[j] = (__bf16)val;
    }
    if (outT) *(uint4*)(outT + (size_t)col * T_TOK + rbase) = u.u;
  }
}

// ---------------------------------------------------------------------------
// 5) G[c] = a_k_c^T @ d_c : [M,64] f32 per chunk.  dT is [64][T] transposed.
__global__ void k_outer(const __bf16* __restrict__ AK, const __bf16* __restrict__ dT,
                        float* __restrict__ G) {
  __shared__ __align__(32) __bf16 sA[4096];   // A[m][i] = ak[i][m]
  __shared__ __align__(32) __bf16 sD[4096];   // B[i][j] = dT[j][i]
  int tid = threadIdx.x, lane = tid & 31, wv = tid >> 5;
  int mb = blockIdx.x * 64, c = blockIdx.y;

  v8f acc[4];
  const v8f vz = V8F_ZERO;
#pragma unroll
  for (int t = 0; t < 4; ++t) acc[t] = vz;

  for (int i0 = 0; i0 < CHUNK; i0 += 64) {
    __syncthreads();
    ldB_tr<2, 4>(sD, dT + (size_t)(c * CHUNK + i0), T_TOK, tid);
    ldA_cm<64, 2>(sA, AK + (size_t)(c * CHUNK + i0) * M_SLOT + mb, M_SLOT, tid);
    wait_async();
    __syncthreads();
#pragma unroll
    for (int nt = 0; nt < 4; ++nt)
#pragma unroll
      for (int kt = 0; kt < 2; ++kt)
        acc[nt] = wmma_bf16(frag_ld(sA, wv * 2 + kt, lane), frag_ld(sD, kt * 4 + nt, lane), acc[nt]);
  }

#pragma unroll
  for (int nt = 0; nt < 4; ++nt) {
    int col   = nt * 16 + (lane & 15);
    int mbase = mb + wv * 16 + ((lane >> 4) << 3);
#pragma unroll
    for (int j = 0; j < 8; ++j)
      G[((size_t)c * M_SLOT + mbase + j) * DD + col] = acc[nt][j];
  }
}

// ---------------------------------------------------------------------------
// 6) exclusive prefix over chunks, emitted TRANSPOSED: ST[c][j][m]
__global__ void k_prefix(const float* __restrict__ G, __bf16* __restrict__ ST) {
  int idx = blockIdx.x * blockDim.x + threadIdx.x;
  if (idx >= M_SLOT * DD) return;
  int m = idx >> 6, j = idx & 63;
  float run = 0.f;
  for (int c = 0; c < NCHUNK; ++c) {
    ST[(size_t)c * (M_SLOT * DD) + (size_t)j * M_SLOT + m] = (__bf16)run;
    run += G[(size_t)c * (M_SLOT * DD) + idx];
  }
}

// ---------------------------------------------------------------------------
// 7) r = u - a_q@S_c - tril_strict(a_q a_k_c^T)@d_c.   NJ = row-block pos +1.
template<int NJ>
__global__ void k_final(const __bf16* __restrict__ AQ, const __bf16* __restrict__ AK,
                        const __bf16* __restrict__ ST, const __bf16* __restrict__ dT,
                        const float* __restrict__ U,   float* __restrict__ R) {
  __shared__ __align__(32) __bf16 smem[4096 * (2 + NJ)];
  __bf16* sAQ = smem;                  // 64x64 A-frag
  __bf16* sS  = smem + 4096;           // 64x64 B-frag
  __bf16* sAK = smem + 8192;           // B^T frag, K=64, N=64*NJ
  __bf16* sW  = smem;                  // overlay: masked 64x64 A-frag
  __bf16* sD  = smem + 4096;           // overlay: 64x64 B-frag

  int tid = threadIdx.x, lane = tid & 31, wv = tid >> 5;
  int c  = blockIdx.x;
  int rb = c * CHUNK + (NJ - 1) * 64;
  const __bf16* STc = ST + (size_t)c * (M_SLOT * DD);

  const v8f vz = V8F_ZERO;
  v8f acc[4];
  v8f w[4 * NJ];
#pragma unroll
  for (int t = 0; t < 4; ++t) acc[t] = vz;
#pragma unroll
  for (int t = 0; t < 4 * NJ; ++t) w[t] = vz;

  for (int m0 = 0; m0 < M_SLOT; m0 += 64) {
    __syncthreads();
    ldA_rm<64, 2>(sAQ, AQ + (size_t)rb * M_SLOT + m0, M_SLOT, tid);
    ldB_tr<2, 4>(sS, STc + m0, M_SLOT, tid);                    // B[m][j]=ST[j][m]
    ldB_tr<2, 4 * NJ>(sAK, AK + (size_t)(c * CHUNK) * M_SLOT + m0, M_SLOT, tid);
    if (m0 + 64 < M_SLOT) {
      __builtin_prefetch(AQ + (size_t)(rb + (tid & 63)) * M_SLOT + m0 + 64, 0, 3);
      __builtin_prefetch(AK + (size_t)(c * CHUNK + tid) * M_SLOT + m0 + 64, 0, 3);
    }
    wait_async();
    __syncthreads();
#pragma unroll
    for (int kt = 0; kt < 2; ++kt) {
      v16bf a = frag_ld(sAQ, wv * 2 + kt, lane);
#pragma unroll
      for (int nt = 0; nt < 4; ++nt)
        acc[nt] = wmma_bf16(a, frag_ld(sS, kt * 4 + nt, lane), acc[nt]);
#pragma unroll
      for (int jt = 0; jt < 4 * NJ; ++jt)
        w[jt] = wmma_bf16(a, frag_ld(sAK, kt * (4 * NJ) + jt, lane), w[jt]);
    }
  }

  // intra-chunk: mask each 64-col block strictly-causal, then @ d block
#pragma unroll
  for (int jb = 0; jb < NJ; ++jb) {
    __syncthreads();
    ldB_tr<2, 4>(sD, dT + (size_t)(c * CHUNK + jb * 64), T_TOK, tid);
#pragma unroll
    for (int nt = 0; nt < 4; ++nt) {
      int lj = nt * 16 + (lane & 15);
      int gj = c * CHUNK + jb * 64 + lj;
      int libase = wv * 16 + ((lane >> 4) << 3);
#pragma unroll
      for (int j = 0; j < 8; ++j) {
        int gi = rb + libase + j;
        sW[fa_idx(libase + j, lj, 2)] = (__bf16)((gj < gi) ? w[jb * 4 + nt][j] : 0.f);
      }
    }
    wait_async();
    __syncthreads();
#pragma unroll
    for (int nt = 0; nt < 4; ++nt)
#pragma unroll
      for (int kt = 0; kt < 2; ++kt)
        acc[nt] = wmma_bf16(frag_ld(sW, wv * 2 + kt, lane), frag_ld(sD, kt * 4 + nt, lane), acc[nt]);
  }

  // r = u - acc
#pragma unroll
  for (int nt = 0; nt < 4; ++nt) {
    int col   = nt * 16 + (lane & 15);
    int rbase = rb + wv * 16 + ((lane >> 4) << 3);
#pragma unroll
    for (int j = 0; j < 8; ++j) {
      size_t o = (size_t)(rbase + j) * DD + col;
      R[o] = U[o] - acc[nt][j];
    }
  }
}

// ---------------------------------------------------------------------------
// workspace layout (bytes)
static const size_t OFF_QB  = 0;
static const size_t OFF_KB  = OFF_QB  + (size_t)T_TOK  * DD * 2;
static const size_t OFF_KMB = OFF_KB  + (size_t)T_TOK  * DD * 2;
static const size_t OFF_VB  = OFF_KMB + (size_t)M_SLOT * DD * 2;
static const size_t OFF_AK  = OFF_VB  + (size_t)M_SLOT * DD * 2;
static const size_t OFF_AQ  = OFF_AK  + (size_t)T_TOK * M_SLOT * 2;
static const size_t OFF_DT  = OFF_AQ  + (size_t)T_TOK * M_SLOT * 2;
static const size_t OFF_UF  = OFF_DT  + (size_t)T_TOK * DD * 2;
static const size_t OFF_G   = OFF_UF  + (size_t)T_TOK * DD * 4;
static const size_t OFF_ST  = OFF_G   + (size_t)NCHUNK * M_SLOT * DD * 4;

extern "C" void kernel_launch(void* const* d_in, const int* in_sizes, int n_in,
                              void* d_out, int out_size, void* d_ws, size_t ws_size,
                              hipStream_t stream) {
  const float* q  = (const float*)d_in[0];
  const float* k  = (const float*)d_in[1];
  const float* v  = (const float*)d_in[2];
  const float* Km = (const float*)d_in[3];
  const float* Vm = (const float*)d_in[4];

  char* ws = (char*)d_ws;
  __bf16* qb  = (__bf16*)(ws + OFF_QB);
  __bf16* kb  = (__bf16*)(ws + OFF_KB);
  __bf16* Kmb = (__bf16*)(ws + OFF_KMB);
  __bf16* Vmb = (__bf16*)(ws + OFF_VB);
  __bf16* ak  = (__bf16*)(ws + OFF_AK);
  __bf16* aq  = (__bf16*)(ws + OFF_AQ);
  __bf16* dT  = (__bf16*)(ws + OFF_DT);
  float*  uF  = (float*) (ws + OFF_UF);
  float*  G   = (float*) (ws + OFF_G);
  __bf16* ST  = (__bf16*)(ws + OFF_ST);
  float*  R   = (float*)d_out;

  const float scale = 0.125f;  // 64^-0.5

  k_cvt_bf16<<<512, 256, 0, stream>>>(q,  qb,  T_TOK  * DD);
  k_cvt_bf16<<<512, 256, 0, stream>>>(k,  kb,  T_TOK  * DD);
  k_cvt_bf16<<<512, 256, 0, stream>>>(Km, Kmb, M_SLOT * DD);
  k_cvt_bf16<<<512, 256, 0, stream>>>(Vm, Vmb, M_SLOT * DD);

  dim3 glog(T_TOK / 64, M_SLOT / 64);
  k_logits<<<glog, 256, 0, stream>>>(kb, Kmb, ak, scale);
  k_logits<<<glog, 256, 0, stream>>>(qb, Kmb, aq, scale);

  k_softmax<<<T_TOK, 256, 0, stream>>>(ak);
  k_softmax<<<T_TOK, 256, 0, stream>>>(aq);

  // d (transposed, bf16) and u
  k_pv<<<T_TOK / 64, 256, 0, stream>>>(ak, Vmb, v, -1.0f, (float*)nullptr, dT);
  k_pv<<<T_TOK / 64, 256, 0, stream>>>(aq, Vmb, (const float*)nullptr, 0.0f, uF, (__bf16*)nullptr);

  dim3 gout(M_SLOT / 64, NCHUNK);
  k_outer<<<gout, 256, 0, stream>>>(ak, dT, G);

  k_prefix<<<(M_SLOT * DD + 255) / 256, 256, 0, stream>>>(G, ST);

  k_final<1><<<NCHUNK, 256, 0, stream>>>(aq, ak, ST, dT, uF, R);
  k_final<2><<<NCHUNK, 256, 0, stream>>>(aq, ak, ST, dT, uF, R);
  k_final<3><<<NCHUNK, 256, 0, stream>>>(aq, ak, ST, dT, uF, R);
  k_final<4><<<NCHUNK, 256, 0, stream>>>(aq, ak, ST, dT, uF, R);
}
